// dec_deeplabv3_contrast_34308198760598
// MI455X (gfx1250) — matrix operand order repaired and checked by reading, not compiled, not run on hardware
//
#include <hip/hip_runtime.h>
#include <stdint.h>

typedef __attribute__((ext_vector_type(8)))  float  v8f;
typedef __attribute__((ext_vector_type(16))) __bf16 v16bf;

#define EPS_BN 1e-5f
#define HW     4096
#define NPIX   16384       // B*H*W = 4*64*64
#define CIN    2048
#define INNER  256
#define NCLS   19
#define QLEN   2975

// ---------------------------------------------------------------- helpers ---
__device__ __forceinline__ unsigned short f2bf(float f) {
  unsigned int u = __float_as_uint(f);
  unsigned int r = (u + 0x7FFFu + ((u >> 16) & 1u)) >> 16;   // RNE
  return (unsigned short)r;
}

struct FragBF { union { uint4 u[2]; v16bf v; }; };
struct Pack8  { union { unsigned short h[8]; uint4 u; }; };

// --------------------------------------------------------- cast / prep -----
__global__ void k_cast_bf16(unsigned short* __restrict__ dst,
                            const float* __restrict__ src, long long n) {
  long long i = (long long)blockIdx.x * blockDim.x + threadIdx.x;
  long long st = (long long)gridDim.x * blockDim.x;
  for (; i < n; i += st) dst[i] = f2bf(src[i]);
}

// OIHW [256][Kin][3][3] fp32 -> per-tap [9][256][Kin] bf16
__global__ void k_prep_tap(unsigned short* __restrict__ dst,
                           const float* __restrict__ src, int Kin, long long n) {
  long long i = (long long)blockIdx.x * blockDim.x + threadIdx.x;
  long long st = (long long)gridDim.x * blockDim.x;
  long long per = (long long)256 * Kin;
  for (; i < n; i += st) {
    int t  = (int)(i / per);
    long long r = i % per;
    int o  = (int)(r / Kin);
    int ci = (int)(r % Kin);
    dst[i] = f2bf(src[((long long)o * Kin + ci) * 9 + t]);
  }
}

// ------------------------------------------------------------ pool branch --
__global__ __launch_bounds__(256) void k_pool_mean(float* __restrict__ xp,
                                                   const float* __restrict__ x) {
  int bc = blockIdx.x;                         // b*2048 + ci
  const float* p = x + (size_t)bc * HW;
  float s = 0.f;
  for (int i = threadIdx.x; i < HW; i += 256) s += p[i];
  __shared__ float red[256];
  red[threadIdx.x] = s; __syncthreads();
  for (int o = 128; o > 0; o >>= 1) {
    if (threadIdx.x < o) red[threadIdx.x] += red[threadIdx.x + o];
    __syncthreads();
  }
  if (threadIdx.x == 0) xp[bc] = red[0] * (1.0f / HW);
}

__global__ __launch_bounds__(256) void k_pool_branch(
    float* __restrict__ pval, const float* __restrict__ w,
    const float* __restrict__ xp, const float* __restrict__ g,
    const float* __restrict__ bt) {
  int co = blockIdx.x;
  __shared__ float red[256];
  __shared__ float pv[4];
  for (int b = 0; b < 4; ++b) {
    float s = 0.f;
    for (int ci = threadIdx.x; ci < CIN; ci += 256)
      s += w[(size_t)co * CIN + ci] * xp[b * CIN + ci];
    red[threadIdx.x] = s; __syncthreads();
    for (int o = 128; o > 0; o >>= 1) {
      if (threadIdx.x < o) red[threadIdx.x] += red[threadIdx.x + o];
      __syncthreads();
    }
    if (threadIdx.x == 0) pv[b] = red[0];
    __syncthreads();
  }
  if (threadIdx.x == 0) {
    float m = 0.25f * (pv[0] + pv[1] + pv[2] + pv[3]);
    float v = 0.f;
    for (int b = 0; b < 4; ++b) { float d = pv[b] - m; v += d * d; }
    v *= 0.25f;
    float r = rsqrtf(v + EPS_BN);
    for (int b = 0; b < 4; ++b)
      pval[co * 4 + b] = fmaxf((pv[b] - m) * r * g[co] + bt[co], 0.0f);
  }
}

__global__ void k_fill_pool(unsigned short* __restrict__ cat0,
                            const float* __restrict__ pval) {
  long long i = (long long)blockIdx.x * blockDim.x + threadIdx.x;
  long long st = (long long)gridDim.x * blockDim.x;
  for (; i < (long long)INNER * NPIX; i += st) {
    int co = (int)(i >> 14);
    int n  = (int)(i & 16383);
    int b  = n >> 12;
    cat0[i] = f2bf(pval[co * 4 + b]);
  }
}

// ------------------------------------------------- WMMA implicit-conv GEMM --
// C[256, 16384] = sum over taps of  A_t[256,K] x B_t[K,16384]
// A in bf16 [taps][256][K]; B element (ci, n) at  b*strideB + ci*strideC + hin*64 + win
#define MT 128
#define NT 64
#define KK 32

__global__ __launch_bounds__(256) void k_gemm(
    float* __restrict__ Y, const unsigned short* __restrict__ Xb,
    const unsigned short* __restrict__ Wb, int K,
    long long strideB, long long strideC, int ntaps, int dil) {
  __shared__ unsigned short As[MT][KK + 8];   // 128 x 40 bf16
  __shared__ unsigned short Bs[NT][KK + 8];   //  64 x 40 bf16
  const int tid   = threadIdx.x;
  const int wave  = tid >> 5;
  const int lane  = tid & 31;
  const int lhalf = lane >> 4;
  const int l16   = lane & 15;
  const int mbase = blockIdx.x * MT;
  const int n0    = blockIdx.y * NT;
  const int b     = n0 >> 12;
  const int h     = (n0 & 4095) >> 6;          // tile spans exactly one image row

  // B-staging assignment: thread owns pixel nl and 8 consecutive k values
  const int nl = tid & 63;
  const int kb = (tid >> 6) * 8;

  v8f cacc[4];
#pragma unroll
  for (int j = 0; j < 4; ++j) cacc[j] = v8f{};

  for (int t = 0; t < ntaps; ++t) {
    int dh = 0, dw = 0;
    if (ntaps != 1) { dh = (t / 3 - 1) * dil; dw = (t % 3 - 1) * dil; }
    const int hin  = h + dh;
    const int win  = nl + dw;
    // single per-thread predicate for this tap; branch-free via clamped address
    const bool ok  = (hin >= 0) && (hin < 64) && (win >= 0) && (win < 64);
    const int hinc = min(max(hin, 0), 63);
    const int winc = min(max(win, 0), 63);
    const long long basePix = (long long)b * strideB + (long long)hinc * 64 + winc;
    const unsigned short* Abase = Wb + (size_t)t * 256 * K;

    for (int k0 = 0; k0 < K; k0 += KK) {
      { // stage A tile 128x32: thread -> row tid/2, 16-elem half tid&1
        int row = tid >> 1;
        int half = tid & 1;
        const unsigned short* g = Abase + (size_t)(mbase + row) * K + k0 + half * 16;
        uint4 a0 = ((const uint4*)g)[0];
        uint4 a1 = ((const uint4*)g)[1];
        if (k0 + KK < K) __builtin_prefetch(g + KK, 0, 1);   // global_prefetch_b8
        uint4* dst = (uint4*)&As[row][half * 16];
        dst[0] = a0; dst[1] = a1;
      }
      { // stage B tile 32x64: 8 strided (ci-major) u16 loads, one ds_store_b128
        Pack8 p;
        const long long a0 = basePix + (long long)(k0 + kb) * strideC;
#pragma unroll
        for (int j = 0; j < 8; ++j) {
          unsigned short v = Xb[a0 + (long long)j * strideC];  // clamped, always legal
          p.h[j] = ok ? v : (unsigned short)0;
        }
        *(uint4*)&Bs[nl][kb] = p.u;
      }
      __syncthreads();

      FragBF a;  // A 16x32 frag: lanes 0-15 row M=l16 K{0..7,16..23}; lanes 16-31 K{8..15,24..31}
      {
        int row = wave * 16 + l16;
        int ko  = lhalf * 8;
        a.u[0] = *(const uint4*)&As[row][ko];
        a.u[1] = *(const uint4*)&As[row][ko + 16];
      }
      const int kob = lhalf * 8;    // B 32x16 frag: lane holds col N=l16, K{kob..kob+7, kob+16..kob+23}
#pragma unroll
      for (int j = 0; j < 4; ++j) {
        FragBF bq;
        int nn = j * 16 + l16;
        bq.u[0] = *(const uint4*)&Bs[nn][kob];
        bq.u[1] = *(const uint4*)&Bs[nn][kob + 16];
        cacc[j] = __builtin_amdgcn_wmma_f32_16x16x32_bf16(
            false, a.v, false, bq.v, (short)0, cacc[j], false, false);
      }
      __syncthreads();
    }
  }
  // epilogue: VGPR v holds M = v + 8*lhalf (ISA C/D layout)
  const int mrow = mbase + wave * 16 + lhalf * 8;
#pragma unroll
  for (int j = 0; j < 4; ++j) {
    int n = n0 + j * 16 + l16;
#pragma unroll
    for (int vv = 0; vv < 8; ++vv)
      Y[(size_t)(mrow + vv) * NPIX + n] = cacc[j][vv];
  }
}

// --------------------------------------------------------------- BN stages --
__global__ __launch_bounds__(256) void k_bnstats(float* __restrict__ stats,
                                                 const float* __restrict__ Y) {
  int co = blockIdx.x;
  const float* p = Y + (size_t)co * NPIX;
  float s = 0.f, s2 = 0.f;
  for (int i = threadIdx.x; i < NPIX; i += 256) { float v = p[i]; s += v; s2 += v * v; }
  __shared__ float rs[256], rq[256];
  rs[threadIdx.x] = s; rq[threadIdx.x] = s2; __syncthreads();
  for (int o = 128; o > 0; o >>= 1) {
    if (threadIdx.x < o) { rs[threadIdx.x] += rs[threadIdx.x + o]; rq[threadIdx.x] += rq[threadIdx.x + o]; }
    __syncthreads();
  }
  if (threadIdx.x == 0) {
    float mean = rs[0] * (1.0f / NPIX);
    float var  = rq[0] * (1.0f / NPIX) - mean * mean;
    stats[co * 2]     = mean;
    stats[co * 2 + 1] = rsqrtf(var + EPS_BN);
  }
}

__global__ void k_bn_cast(unsigned short* __restrict__ dst, const float* __restrict__ Y,
                          const float* __restrict__ stats, const float* __restrict__ g,
                          const float* __restrict__ bt) {
  long long i = (long long)blockIdx.x * blockDim.x + threadIdx.x;
  long long st = (long long)gridDim.x * blockDim.x;
  for (; i < (long long)INNER * NPIX; i += st) {
    int co = (int)(i >> 14);
    float v = (Y[i] - stats[co * 2]) * stats[co * 2 + 1] * g[co] + bt[co];
    dst[i] = f2bf(fmaxf(v, 0.0f));
  }
}

__global__ void k_bn_f32(float* __restrict__ dst, const float* __restrict__ Y,
                         const float* __restrict__ stats, const float* __restrict__ g,
                         const float* __restrict__ bt) {
  long long i = (long long)blockIdx.x * blockDim.x + threadIdx.x;
  long long st = (long long)gridDim.x * blockDim.x;
  for (; i < (long long)INNER * NPIX; i += st) {
    int co = (int)(i >> 14);
    float v = (Y[i] - stats[co * 2]) * stats[co * 2 + 1] * g[co] + bt[co];
    dst[i] = fmaxf(v, 0.0f);
  }
}

// ----------------------------------------------- final conv + argmax --------
__global__ __launch_bounds__(256) void k_final(float* __restrict__ res, int* __restrict__ pred,
                                               const float* __restrict__ fea,
                                               const float* __restrict__ fw,
                                               const float* __restrict__ fb) {
  __shared__ float sfw[NCLS * INNER];
  for (int i = threadIdx.x; i < NCLS * INNER; i += 256) sfw[i] = fw[i];
  __syncthreads();
  int n = blockIdx.x * 256 + threadIdx.x;
  float acc[NCLS];
#pragma unroll
  for (int c = 0; c < NCLS; ++c) acc[c] = fb[c];
  for (int co = 0; co < INNER; ++co) {
    float f = fea[(size_t)co * NPIX + n];
#pragma unroll
    for (int c = 0; c < NCLS; ++c) acc[c] += sfw[c * INNER + co] * f;
  }
  int b = n >> 12, hw = n & 4095;
  float best = acc[0]; int bi = 0;
#pragma unroll
  for (int c = 0; c < NCLS; ++c) {
    res[((size_t)b * NCLS + c) * HW + hw] = acc[c];
    if (acc[c] > best) { best = acc[c]; bi = c; }
  }
  pred[n] = bi;
}

// ------------------------------- deterministic segment reductions / keys ---
__global__ __launch_bounds__(256) void k_counts(float* __restrict__ counts,
                                                const int* __restrict__ pred) {
  int c = blockIdx.x;
  float s = 0.f;
  for (int n = threadIdx.x; n < NPIX; n += 256) s += (pred[n] == c) ? 1.0f : 0.0f;
  __shared__ float red[256];
  red[threadIdx.x] = s; __syncthreads();
  for (int o = 128; o > 0; o >>= 1) {
    if (threadIdx.x < o) red[threadIdx.x] += red[threadIdx.x + o];
    __syncthreads();
  }
  if (threadIdx.x == 0) counts[c] = red[0];
}

__global__ __launch_bounds__(256) void k_segsum(float* __restrict__ sums,
                                                const float* __restrict__ fea,
                                                const int* __restrict__ pred) {
  int co = blockIdx.x, c = blockIdx.y;
  float s = 0.f;
  for (int n = threadIdx.x; n < NPIX; n += 256)
    s += (pred[n] == c) ? fea[(size_t)co * NPIX + n] : 0.0f;
  __shared__ float red[256];
  red[threadIdx.x] = s; __syncthreads();
  for (int o = 128; o > 0; o >>= 1) {
    if (threadIdx.x < o) red[threadIdx.x] += red[threadIdx.x + o];
    __syncthreads();
  }
  if (threadIdx.x == 0) sums[c * INNER + co] = red[0];
}

__global__ __launch_bounds__(256) void k_keys(float* __restrict__ keys,
                                              const float* __restrict__ sums,
                                              const float* __restrict__ counts) {
  int c = blockIdx.x, co = threadIdx.x;
  float mean = sums[c * INNER + co] / fmaxf(counts[c], 1.0f);
  __shared__ float red[256];
  __shared__ float nrm;
  red[co] = mean * mean; __syncthreads();
  for (int o = 128; o > 0; o >>= 1) {
    if (co < o) red[co] += red[co + o];
    __syncthreads();
  }
  if (co == 0) nrm = fmaxf(sqrtf(red[0]), 1e-12f);
  __syncthreads();
  keys[c * INNER + co] = mean / nrm;
}

// -------------------------------------------------- contrastive loss -------
__global__ void k_qsum(float* __restrict__ qsum, const float* __restrict__ q) {
  long long i = (long long)blockIdx.x * blockDim.x + threadIdx.x;
  long long st = (long long)gridDim.x * blockDim.x;
  for (; i < (long long)INNER * QLEN; i += st) {
    float s = 0.f;
    for (int c = 0; c < NCLS; ++c) s += q[(size_t)c * INNER * QLEN + i];
    qsum[i] = s;
  }
}

__global__ __launch_bounds__(256) void k_contrast(float* __restrict__ ce,
                                                  const float* __restrict__ keys,
                                                  const float* __restrict__ q,
                                                  const float* __restrict__ qsum) {
  int c = blockIdx.x, r = threadIdx.x;
  const float s = keys[c * INNER + r] * 5.0f;   // 1 / TEMP
  const float* qc = q + ((size_t)c * INNER + r) * QLEN;
  const float* qs = qsum + (size_t)r * QLEN;
  float mx = -3.4e38f;
  for (int j = 0; j < QLEN; ++j) {
    float lp = s * qc[j];
    float ln = s * (qs[j] - qc[j]);
    mx = fmaxf(mx, fmaxf(lp, ln));
  }
  float sum = 0.f;
  for (int j = 0; j < QLEN; ++j) {
    float lp = s * qc[j];
    float ln = s * (qs[j] - qc[j]);
    sum += expf(lp - mx) + expf(ln - mx);
  }
  float val = mx + logf(sum) - s * qc[0];
  __shared__ float red[256];
  red[r] = val; __syncthreads();
  for (int o = 128; o > 0; o >>= 1) {
    if (r < o) red[r] += red[r + o];
    __syncthreads();
  }
  if (r == 0) ce[c] = red[0] * (1.0f / INNER);
}

__global__ void k_sumce(float* __restrict__ loss, const float* __restrict__ ce,
                        const float* __restrict__ counts) {
  if (threadIdx.x == 0) {
    float s = 0.f;
    for (int c = 0; c < NCLS; ++c) if (counts[c] > 0.0f) s += ce[c];
    loss[0] = s;
  }
}

// ---------------------------------------------------------------- launch ---
extern "C" void kernel_launch(void* const* d_in, const int* in_sizes, int n_in,
                              void* d_out, int out_size, void* d_ws, size_t ws_size,
                              hipStream_t stream) {
  const float* x      = (const float*)d_in[0];
  const float* w_pool = (const float*)d_in[1];
  const float* g_pool = (const float*)d_in[2];
  const float* b_pool = (const float*)d_in[3];
  const float* w1 = (const float*)d_in[4];
  const float* g1 = (const float*)d_in[5];
  const float* b1 = (const float*)d_in[6];
  const float* w2 = (const float*)d_in[7];
  const float* g2 = (const float*)d_in[8];
  const float* b2 = (const float*)d_in[9];
  const float* w3 = (const float*)d_in[10];
  const float* g3 = (const float*)d_in[11];
  const float* b3 = (const float*)d_in[12];
  const float* w4 = (const float*)d_in[13];
  const float* g4 = (const float*)d_in[14];
  const float* b4 = (const float*)d_in[15];
  const float* wh = (const float*)d_in[16];
  const float* gh = (const float*)d_in[17];
  const float* bh = (const float*)d_in[18];
  const float* fw = (const float*)d_in[19];
  const float* fb = (const float*)d_in[20];
  const float* qu = (const float*)d_in[21];
  float* out = (float*)d_out;

  // workspace carve-up (256-byte aligned)
  char* base = (char*)d_ws;
  size_t off = 0;
  auto carve = [&](size_t bytes) -> void* {
    void* p = base + off;
    off += (bytes + 255) & ~(size_t)255;
    return p;
  };
  unsigned short* xb   = (unsigned short*)carve((size_t)4 * CIN * HW * 2);      // bf16 x
  unsigned short* w1b  = (unsigned short*)carve((size_t)INNER * CIN * 2);
  unsigned short* w2b  = (unsigned short*)carve((size_t)9 * INNER * CIN * 2);
  unsigned short* w3b  = (unsigned short*)carve((size_t)9 * INNER * CIN * 2);
  unsigned short* w4b  = (unsigned short*)carve((size_t)9 * INNER * CIN * 2);
  unsigned short* whb  = (unsigned short*)carve((size_t)9 * INNER * 1280 * 2);
  unsigned short* cat  = (unsigned short*)carve((size_t)1280 * NPIX * 2);       // post-BN bf16
  float* Ytmp   = (float*)carve((size_t)INNER * NPIX * 4);
  float* fea    = (float*)carve((size_t)INNER * NPIX * 4);
  float* xp     = (float*)carve((size_t)4 * CIN * 4);
  float* pval   = (float*)carve((size_t)INNER * 4 * 4);
  float* stats  = (float*)carve((size_t)INNER * 2 * 4);
  int*   pred   = (int*)carve((size_t)NPIX * 4);
  float* sums   = (float*)carve((size_t)NCLS * INNER * 4);
  float* counts = (float*)carve((size_t)NCLS * 4);
  float* keys   = (float*)carve((size_t)NCLS * INNER * 4);
  float* ce     = (float*)carve((size_t)NCLS * 4);
  float* qsum   = (float*)carve((size_t)INNER * QLEN * 4);

  // 1) casts / weight prep
  k_cast_bf16<<<4096, 256, 0, stream>>>(xb, x, (long long)4 * CIN * HW);
  k_cast_bf16<<<2048, 256, 0, stream>>>(w1b, w1, (long long)INNER * CIN);
  k_prep_tap<<<4096, 256, 0, stream>>>(w2b, w2, CIN, (long long)9 * INNER * CIN);
  k_prep_tap<<<4096, 256, 0, stream>>>(w3b, w3, CIN, (long long)9 * INNER * CIN);
  k_prep_tap<<<4096, 256, 0, stream>>>(w4b, w4, CIN, (long long)9 * INNER * CIN);
  k_prep_tap<<<4096, 256, 0, stream>>>(whb, wh, 1280, (long long)9 * INNER * 1280);

  // 2) image-pool branch (fp32, tiny)
  k_pool_mean<<<4 * CIN, 256, 0, stream>>>(xp, x);
  k_pool_branch<<<INNER, 256, 0, stream>>>(pval, w_pool, xp, g_pool, b_pool);
  k_fill_pool<<<4096, 256, 0, stream>>>(cat, pval);   // branch 0 of concat

  // 3) ASPP conv branches via bf16 WMMA implicit GEMM
  const long long SBx = (long long)CIN * HW;   // x: NCHW  b-stride
  const long long SCx = HW;                    // x: ci-stride
  dim3 gg(INNER / MT, NPIX / NT);              // (2, 256)
  struct { const unsigned short* w; const float *g, *b; int taps, dil; } br[4] = {
      {w1b, g1, b1, 1, 1}, {w2b, g2, b2, 9, 12}, {w3b, g3, b3, 9, 24}, {w4b, g4, b4, 9, 36}};
  for (int i = 0; i < 4; ++i) {
    k_gemm<<<gg, 256, 0, stream>>>(Ytmp, xb, br[i].w, CIN, SBx, SCx, br[i].taps, br[i].dil);
    k_bnstats<<<INNER, 256, 0, stream>>>(stats, Ytmp);
    k_bn_cast<<<4096, 256, 0, stream>>>(cat + (size_t)(i + 1) * INNER * NPIX,
                                        Ytmp, stats, br[i].g, br[i].b);
  }

  // 4) head 3x3 conv over 1280-channel concat (bf16 WMMA), BN -> fp32 fea
  k_gemm<<<gg, 256, 0, stream>>>(Ytmp, cat, whb, 1280, (long long)HW, (long long)NPIX, 9, 1);
  k_bnstats<<<INNER, 256, 0, stream>>>(stats, Ytmp);
  k_bn_f32<<<4096, 256, 0, stream>>>(fea, Ytmp, stats, gh, bh);

  // 5) final 1x1 conv + argmax (res -> d_out)
  k_final<<<NPIX / 256, 256, 0, stream>>>(out, pred, fea, fw, fb);

  // 6) deterministic per-class means and keys
  k_counts<<<NCLS, 256, 0, stream>>>(counts, pred);
  dim3 gs(INNER, NCLS);
  k_segsum<<<gs, 256, 0, stream>>>(sums, fea, pred);
  k_keys<<<NCLS, 256, 0, stream>>>(keys, sums, counts);

  // 7) contrastive loss
  k_qsum<<<2975, 256, 0, stream>>>(qsum, qu);
  k_contrast<<<NCLS, 256, 0, stream>>>(ce, keys, qu, qsum);
  k_sumce<<<1, 32, 0, stream>>>(out + (out_size - 1), ce, counts);
}
